// DGMMC_diagonal_54185307406410
// MI455X (gfx1250) — compile-verified, matrix operand order checked by reading, not run on hardware
//
#include <hip/hip_runtime.h>
#include <math.h>

#define NC   200
#define NK   16
#define ND   512
#define NB   8192
#define NCK  3200          // NC*NK
#define KAUG 1024          // 2*ND
#define LOG_2PI 1.8378770664093453f

typedef __attribute__((ext_vector_type(16))) __bf16 v16bf;
typedef __attribute__((ext_vector_type(8)))  float  v8f;
typedef __attribute__((ext_vector_type(4)))  int    v4i;

union Frag { v16bf v; uint4 q[2]; };

#define AS1 __attribute__((address_space(1)))
#define AS3 __attribute__((address_space(3)))

// ---------- bf16 split helpers (round-to-nearest-even) ----------
__device__ inline unsigned short f32_bf16_rne(float f) {
  unsigned int u = __float_as_uint(f);
  u += 0x7FFFu + ((u >> 16) & 1u);
  return (unsigned short)(u >> 16);
}
__device__ inline float bf16_f32(unsigned short h) {
  return __uint_as_float(((unsigned int)h) << 16);
}
__device__ inline void split2(float v, unsigned short* hi, unsigned short* lo) {
  unsigned short h = f32_bf16_rne(v);
  *hi = h;
  *lo = f32_bf16_rne(v - bf16_f32(h));
}

// ---------- kernel 1: log(softmax(priors)+1e-6) ----------
__global__ void prep_logp(const float* __restrict__ priors, float* __restrict__ logp) {
  __shared__ float sm[256];
  int t = threadIdx.x;
  float v = (t < NC) ? priors[t] : -INFINITY;
  sm[t] = v; __syncthreads();
  for (int s = 128; s > 0; s >>= 1) { if (t < s) sm[t] = fmaxf(sm[t], sm[t + s]); __syncthreads(); }
  float mx = sm[0]; __syncthreads();
  float e = (t < NC) ? __expf(v - mx) : 0.f;
  sm[t] = e; __syncthreads();
  for (int s = 128; s > 0; s >>= 1) { if (t < s) sm[t] += sm[t + s]; __syncthreads(); }
  if (t < NC) logp[t] = __logf(e / sm[0] + 1e-6f);
}

// ---------- kernel 2: per-component B matrix (bf16 hi/lo) + fused constants ----------
__global__ void prep_comp(const float* __restrict__ means, const float* __restrict__ bw,
                          const float* __restrict__ w,
                          unsigned short* __restrict__ Bh, unsigned short* __restrict__ Bl,
                          float* __restrict__ constj) {
  __shared__ float sld[256], scc[256];
  int j = blockIdx.x;
  int t = threadIdx.x;
  float ld = 0.f, cc = 0.f;
  for (int d = t; d < ND; d += 256) {
    float b   = fminf(fmaxf(bw[(size_t)j * ND + d], 1e-6f), 1000.f);
    float inv = 1.f / b;
    float m   = means[(size_t)j * ND + d];
    unsigned short h, l;
    split2(-0.5f * inv, &h, &l);                 // pairs with x^2
    Bh[(size_t)j * KAUG + d] = h; Bl[(size_t)j * KAUG + d] = l;
    split2(m * inv, &h, &l);                     // pairs with x  (−0.5·−2 = +1)
    Bh[(size_t)j * KAUG + ND + d] = h; Bl[(size_t)j * KAUG + ND + d] = l;
    ld += __logf(b);
    cc += m * m * inv;
  }
  sld[t] = ld; scc[t] = cc; __syncthreads();
  for (int s = 128; s > 0; s >>= 1) {
    if (t < s) { sld[t] += sld[t + s]; scc[t] += scc[t + s]; }
    __syncthreads();
  }
  if (t == 0) {
    int c = j >> 4;                              // class of this component
    float mx = -INFINITY;
    for (int k = 0; k < NK; k++) mx = fmaxf(mx, w[c * NK + k]);
    float s = 0.f;
    for (int k = 0; k < NK; k++) s += __expf(w[c * NK + k] - mx);
    float lw = __logf(__expf(w[j] - mx) / s + 1e-6f);
    constj[j] = -0.5f * ((float)ND * LOG_2PI + sld[0] + scc[0]) + lw;
  }
}

// ---------- kernel 3: augmented X = [x^2, x] as bf16 hi/lo ----------
__global__ void prep_x(const float* __restrict__ x,
                       unsigned short* __restrict__ Ah, unsigned short* __restrict__ Al) {
  int i = blockIdx.x * blockDim.x + threadIdx.x;
  if (i >= NB * ND) return;
  int row = i >> 9, d = i & (ND - 1);
  float v = x[i];
  unsigned short h, l;
  split2(v * v, &h, &l);
  Ah[(size_t)row * KAUG + d] = h;      Al[(size_t)row * KAUG + d] = l;
  split2(v, &h, &l);
  Ah[(size_t)row * KAUG + ND + d] = h; Al[(size_t)row * KAUG + ND + d] = l;
}

// ---------- async 16B global -> LDS copy (CDNA5 ASYNCcnt path) ----------
__device__ inline void async16(const unsigned short* g, char* lds) {
  __builtin_amdgcn_global_load_async_to_lds_b128(
      (AS1 v4i*)g, (AS3 v4i*)lds, 0, 0);
}

// ---------- kernel 4: WMMA GEMM (bf16x3 split), async double-buffered LDS ----------
// Block tile 128x128, 8 wave32 as 4(M) x 2(N); each wave computes 32x64.
// LDS per buffer: A_hi 8K | A_lo 8K | B_hi 8K | B_lo 8K = 32KB; x2 buffers = 64KB.
#define LDS_AHI 0
#define LDS_ALO 8192
#define LDS_BHI 16384
#define LDS_BLO 24576
#define LDS_BUF 32768

__global__ __launch_bounds__(256)
void gmm_wmma(const unsigned short* __restrict__ Ah, const unsigned short* __restrict__ Al,
              const unsigned short* __restrict__ Bh, const unsigned short* __restrict__ Bl,
              const float* __restrict__ constj, float* __restrict__ lse1) {
  __shared__ char smem[2 * LDS_BUF];

  const int lane = threadIdx.x & 31;
  const int wave = threadIdx.x >> 5;
  const int wm = wave & 3;                  // 4 waves along M
  const int wn = wave >> 2;                 // 2 waves along N
  const int rowBlk = blockIdx.y * 128;
  const int colBlk = blockIdx.x * 128;
  const int r  = lane & 15;                 // row within 16-frag
  const int kh = (lane >> 4) << 3;          // K-half: 0 or 8 (ISA 16-bit operand layout)

  // async-copy assignment: each wave moves 2 x 512B segments of each plane.
  // segment t = 8 rows (stride 2) of 64B; lane -> (row = 2*(l>>2)+t, 16B chunk = l&3)
  const int cpRow0 = wave * 16 + (lane >> 2) * 2;   // rows wave*16 .. wave*16+15 (2 segs)
  const int cpChunk = lane & 3;

  const v8f vzero = {0.f, 0.f, 0.f, 0.f, 0.f, 0.f, 0.f, 0.f};
  v8f acc[2][4];
#pragma unroll
  for (int mi = 0; mi < 2; mi++)
#pragma unroll
    for (int ni = 0; ni < 4; ni++) acc[mi][ni] = vzero;

  // ---- issue async stage of k-block kb into buffer buf ----
  auto stage = [&](int kk, int buf) {
    char* base = smem + buf * LDS_BUF;
#pragma unroll
    for (int t = 0; t < 2; t++) {
      int row = cpRow0 + t;                          // 0..127
      size_t gOff = (size_t)row * KAUG + kk + cpChunk * 8;
      int    lOff = row * 64 + cpChunk * 16;
      async16(Ah + (size_t)rowBlk * KAUG + gOff, base + LDS_AHI + lOff);
      async16(Al + (size_t)rowBlk * KAUG + gOff, base + LDS_ALO + lOff);
      async16(Bh + (size_t)colBlk * KAUG + gOff, base + LDS_BHI + lOff);
      async16(Bl + (size_t)colBlk * KAUG + gOff, base + LDS_BLO + lOff);
    }
  };

  stage(0, 0);                                       // prologue: fill buffer 0

  const int aRowOff = (wm * 32 + r) * 64 + kh * 2;   // bytes into A plane (mi adds 16*64)
  const int bRowOff = (wn * 64 + r) * 64 + kh * 2;   // bytes into B plane (ni adds 16*64)

  for (int kb = 0; kb < KAUG / 32; kb++) {
    int cur = kb & 1;
    if ((kb + 1) < (KAUG / 32)) {
      stage((kb + 1) * 32, cur ^ 1);                 // prefetch next buffer (async)
      __builtin_amdgcn_s_wait_asynccnt(8);           // my writes to buf[cur] landed
    } else {
      __builtin_amdgcn_s_wait_asynccnt(0);
    }
    __syncthreads();                                 // everyone's writes visible

    char* base = smem + cur * LDS_BUF;
    Frag fa_h[2], fa_l[2], fb_h[4], fb_l[4];
#pragma unroll
    for (int mi = 0; mi < 2; mi++) {
      int o = aRowOff + mi * 16 * 64;
      fa_h[mi].q[0] = *(const uint4*)(base + LDS_AHI + o);
      fa_h[mi].q[1] = *(const uint4*)(base + LDS_AHI + o + 32);
      fa_l[mi].q[0] = *(const uint4*)(base + LDS_ALO + o);
      fa_l[mi].q[1] = *(const uint4*)(base + LDS_ALO + o + 32);
    }
#pragma unroll
    for (int ni = 0; ni < 4; ni++) {
      int o = bRowOff + ni * 16 * 64;
      fb_h[ni].q[0] = *(const uint4*)(base + LDS_BHI + o);
      fb_h[ni].q[1] = *(const uint4*)(base + LDS_BHI + o + 32);
      fb_l[ni].q[0] = *(const uint4*)(base + LDS_BLO + o);
      fb_l[ni].q[1] = *(const uint4*)(base + LDS_BLO + o + 32);
    }
#pragma unroll
    for (int mi = 0; mi < 2; mi++)
#pragma unroll
      for (int ni = 0; ni < 4; ni++) {
        acc[mi][ni] = __builtin_amdgcn_wmma_f32_16x16x32_bf16(
            false, fa_h[mi].v, false, fb_h[ni].v, (short)0, acc[mi][ni], false, false);
        acc[mi][ni] = __builtin_amdgcn_wmma_f32_16x16x32_bf16(
            false, fa_h[mi].v, false, fb_l[ni].v, (short)0, acc[mi][ni], false, false);
        acc[mi][ni] = __builtin_amdgcn_wmma_f32_16x16x32_bf16(
            false, fa_l[mi].v, false, fb_h[ni].v, (short)0, acc[mi][ni], false, false);
      }

    __syncthreads();                                 // all reads of buf[cur] done
  }

  // Epilogue: log_prob = acc + const[col]; logsumexp over the 16 components of a
  // class == reduction across the 16 lanes of each wave half (C-matrix layout).
  const int nlane = lane & 15;
  const int half  = lane >> 4;
#pragma unroll
  for (int mi = 0; mi < 2; mi++)
#pragma unroll
    for (int ni = 0; ni < 4; ni++) {
      int col = colBlk + wn * 64 + ni * 16 + nlane;
      float cj = constj[col];
      int cls = (colBlk + wn * 64 + ni * 16) >> 4;
#pragma unroll
      for (int v = 0; v < 8; v++) {
        float val = acc[mi][ni][v] + cj;
        float m = val;
#pragma unroll
        for (int off = 1; off < 16; off <<= 1) m = fmaxf(m, __shfl_xor(m, off, 32));
        float e = __expf(val - m);
#pragma unroll
        for (int off = 1; off < 16; off <<= 1) e += __shfl_xor(e, off, 32);
        if (nlane == 0) {
          int row = rowBlk + wm * 32 + mi * 16 + v + (half << 3);
          lse1[(size_t)row * NC + cls] = m + __logf(e);
        }
      }
    }
}

// ---------- kernel 5: + log p, normalize over classes ----------
__global__ void finalize_norm(const float* __restrict__ lse1, const float* __restrict__ logp,
                              float* __restrict__ out) {
  __shared__ float sm[256];
  int row = blockIdx.x, t = threadIdx.x;
  float v = -INFINITY;
  if (t < NC) v = lse1[(size_t)row * NC + t] + logp[t];
  sm[t] = v; __syncthreads();
  for (int s = 128; s > 0; s >>= 1) { if (t < s) sm[t] = fmaxf(sm[t], sm[t + s]); __syncthreads(); }
  float mx = sm[0]; __syncthreads();
  float e = (t < NC) ? __expf(v - mx) : 0.f;
  sm[t] = e; __syncthreads();
  for (int s = 128; s > 0; s >>= 1) { if (t < s) sm[t] += sm[t + s]; __syncthreads(); }
  float lse = mx + __logf(sm[0]);
  if (t < NC) out[(size_t)row * NC + t] = v - lse;
}

extern "C" void kernel_launch(void* const* d_in, const int* in_sizes, int n_in,
                              void* d_out, int out_size, void* d_ws, size_t ws_size,
                              hipStream_t stream) {
  const float* x      = (const float*)d_in[0];   // [8192, 512]
  const float* means  = (const float*)d_in[1];   // [3200, 512]
  const float* bw     = (const float*)d_in[2];   // [3200, 512]
  const float* w      = (const float*)d_in[3];   // [3200]
  const float* priors = (const float*)d_in[4];   // [200]
  float* out = (float*)d_out;                    // [8192, 200]

  char* ws = (char*)d_ws;
  const size_t szA = (size_t)NB  * KAUG * sizeof(unsigned short);  // 16 MB each
  const size_t szB = (size_t)NCK * KAUG * sizeof(unsigned short);  // 6.5 MB each
  unsigned short* Ah = (unsigned short*)(ws);
  unsigned short* Al = (unsigned short*)(ws + szA);
  unsigned short* Bh = (unsigned short*)(ws + 2 * szA);
  unsigned short* Bl = (unsigned short*)(ws + 2 * szA + szB);
  float* constj = (float*)(ws + 2 * szA + 2 * szB);
  float* logp   = constj + NCK;
  float* lse1   = logp + 256;                                      // padded

  prep_logp<<<1, 256, 0, stream>>>(priors, logp);
  prep_comp<<<NCK, 256, 0, stream>>>(means, bw, w, Bh, Bl, constj);
  prep_x<<<(NB * ND + 255) / 256, 256, 0, stream>>>(x, Ah, Al);

  dim3 grid(NCK / 128, NB / 128);   // 25 x 64 workgroups
  gmm_wmma<<<grid, 256, 0, stream>>>(Ah, Al, Bh, Bl, constj, lse1);

  finalize_norm<<<NB, 256, 0, stream>>>(lse1, logp, out);
}
